// GraphConvLayer_57251914056302
// MI455X (gfx1250) — matrix operand order repaired and checked by reading, not compiled, not run on hardware
//
#include <hip/hip_runtime.h>
#include <hip/hip_bf16.h>

// CDNA5 / gfx1250, wave32. Matrix work via v_wmma_f32_16x16x32_f16
// (f16 operands, f32 accumulate); everything else f32.

typedef __attribute__((ext_vector_type(16))) _Float16 v16h;
typedef __attribute__((ext_vector_type(8)))  _Float16 v8h;
typedef __attribute__((ext_vector_type(8)))  float    v8f;

// ---------------------------------------------------------------------------
// helpers
// ---------------------------------------------------------------------------
__device__ __forceinline__ v16h pack16(float4 x0, float4 x1, float4 x2, float4 x3) {
  v16h a;
  a[0]=(_Float16)x0.x; a[1]=(_Float16)x0.y; a[2]=(_Float16)x0.z; a[3]=(_Float16)x0.w;
  a[4]=(_Float16)x1.x; a[5]=(_Float16)x1.y; a[6]=(_Float16)x1.z; a[7]=(_Float16)x1.w;
  a[8]=(_Float16)x2.x; a[9]=(_Float16)x2.y; a[10]=(_Float16)x2.z; a[11]=(_Float16)x2.w;
  a[12]=(_Float16)x3.x; a[13]=(_Float16)x3.y; a[14]=(_Float16)x3.z; a[15]=(_Float16)x3.w;
  return a;
}

// Stage a [K x 128] f32 weight matrix into LDS as f16 B-fragments.
// Fragment (c,n) covers K rows [32c,32c+32), N cols [16n,16n+16).
// B VGPR layout (16-bit, 32x16): lanes 0-15 hold K=0..7,16..23 (N=lane),
// lanes 16-31 hold K=8..15,24..31 (N=lane-16); per-lane 16 halves contiguous.
__device__ __forceinline__ void stage_wfrags(const float* __restrict__ W,
                                             _Float16* __restrict__ dst,
                                             int nFrags /* = (K/32)*8 */) {
  for (int i = threadIdx.x; i < nFrags * 512; i += blockDim.x) {
    int frag = i >> 9, rem = i & 511;
    int l = rem >> 4, j = rem & 15;
    int c = frag >> 3, n = frag & 7;
    int K = c * 32 + ((l & 16) ? 8 : 0) + (j < 8 ? j : j + 8);
    int N = n * 16 + (l & 15);
    dst[i] = (_Float16)W[K * 128 + N];
  }
}

// ---------------------------------------------------------------------------
// Kernel 0: zero the aggregation buffer
// ---------------------------------------------------------------------------
__global__ void __launch_bounds__(256)
zero_kernel(float4* __restrict__ p, long long n4) {
  long long stride = (long long)gridDim.x * blockDim.x;
  for (long long i = (long long)blockIdx.x * blockDim.x + threadIdx.x; i < n4; i += stride)
    p[i] = make_float4(0.f, 0.f, 0.f, 0.f);
}

// ---------------------------------------------------------------------------
// Kernel 1: edge message MLP + attention logits.
// One wave = 16 edges. msg_in = [nf[row] | nf[col] | ef] (K=288, 9 chunks).
// ---------------------------------------------------------------------------
__global__ void __launch_bounds__(256)
edge_msg_kernel(const float* __restrict__ nf, const float* __restrict__ ef,
                const float* __restrict__ mW1, const float* __restrict__ mb1,
                const float* __restrict__ mW2, const float* __restrict__ mb2,
                const float* __restrict__ aW,  const float* __restrict__ ab,
                const int* __restrict__ ei,
                _Float16* __restrict__ msg, float* __restrict__ logits,
                int E, int nTiles) {
  extern __shared__ char smem[];
  _Float16* w1f = (_Float16*)smem;          // 72 frags * 512 halves
  _Float16* w2f = w1f + 72 * 512;           // 32 frags * 512 halves
  _Float16* scratchAll = w2f + 32 * 512;    // 8 waves * 2048 halves
  float* aWs = (float*)(scratchAll + 8 * 2048);
  float* b1s = aWs + 128;
  float* b2s = b1s + 128;

  stage_wfrags(mW1, w1f, 72);
  stage_wfrags(mW2, w2f, 32);
  for (int i = threadIdx.x; i < 128; i += blockDim.x) {
    aWs[i] = aW[i]; b1s[i] = mb1[i]; b2s[i] = mb2[i];
  }
  __syncthreads();

  const int lane  = threadIdx.x & 31;
  const int wave  = threadIdx.x >> 5;
  const int laneN = lane & 15;
  const int koff  = (lane < 16) ? 0 : 8;   // A layout half-K offset
  const int mBase = (lane < 16) ? 0 : 8;   // C/D layout row offset
  _Float16* hl = scratchAll + wave * 2048; // 16x128 f16 per-wave scratch
  const float abv = ab[0];

  int wgid = blockIdx.x * (blockDim.x >> 5) + wave;
  int wstride = gridDim.x * (blockDim.x >> 5);
  for (int t = wgid; t < nTiles; t += wstride) {
    const int eBase = t << 4;
    const int e = eBase + laneN;
    const int ridx = ei[e];
    const int cidx = ei[E + e];

    // ---- GEMM 1: [16,288] x [288,128], bias in accumulator ----
    v8f acc[8];
#pragma unroll
    for (int n = 0; n < 8; ++n) {
      float bv = b1s[n * 16 + laneN];
#pragma unroll
      for (int r = 0; r < 8; ++r) acc[n][r] = bv;
    }
#pragma unroll
    for (int c = 0; c < 9; ++c) {
      const float* p;
      if (c < 4)      p = nf + (size_t)ridx * 128 + c * 32 + koff;
      else if (c < 8) p = nf + (size_t)cidx * 128 + (c - 4) * 32 + koff;
      else            p = ef + (size_t)e * 32 + koff;
      float4 x0 = *(const float4*)p;
      float4 x1 = *(const float4*)(p + 4);
      float4 x2 = *(const float4*)(p + 16);
      float4 x3 = *(const float4*)(p + 20);
      v16h a = pack16(x0, x1, x2, x3);
#pragma unroll
      for (int n = 0; n < 8; ++n) {
        v16h b = *(const v16h*)(w1f + ((c * 8 + n) << 9) + (lane << 4));
        acc[n] = __builtin_amdgcn_wmma_f32_16x16x32_f16(
            false, a, false, b, (short)0, acc[n], false, false);
      }
    }

    // ---- ReLU, re-layout h (D layout -> A layout) through LDS ----
#pragma unroll
    for (int n = 0; n < 8; ++n)
#pragma unroll
      for (int r = 0; r < 8; ++r) {
        float v = acc[n][r];
        hl[(r + mBase) * 128 + n * 16 + laneN] = (_Float16)(v > 0.f ? v : 0.f);
      }

    // ---- GEMM 2: [16,128] x [128,128] ----
    v8f acc2[8];
#pragma unroll
    for (int n = 0; n < 8; ++n) {
      float bv = b2s[n * 16 + laneN];
#pragma unroll
      for (int r = 0; r < 8; ++r) acc2[n][r] = bv;
    }
#pragma unroll
    for (int c = 0; c < 4; ++c) {
      v8h lo = *(const v8h*)(hl + laneN * 128 + c * 32 + koff);
      v8h hi = *(const v8h*)(hl + laneN * 128 + c * 32 + koff + 16);
      v16h a = __builtin_shufflevector(lo, hi, 0,1,2,3,4,5,6,7,8,9,10,11,12,13,14,15);
#pragma unroll
      for (int n = 0; n < 8; ++n) {
        v16h b = *(const v16h*)(w2f + ((c * 8 + n) << 9) + (lane << 4));
        acc2[n] = __builtin_amdgcn_wmma_f32_16x16x32_f16(
            false, a, false, b, (short)0, acc2[n], false, false);
      }
    }

    // ---- attention logit: messages @ a_W + a_b (cross-lane reduce) ----
#pragma unroll
    for (int r = 0; r < 8; ++r) {
      float s = 0.f;
#pragma unroll
      for (int n = 0; n < 8; ++n) s += acc2[n][r] * aWs[n * 16 + laneN];
#pragma unroll
      for (int m = 1; m <= 8; m <<= 1) s += __shfl_xor(s, m, 32);
      if (laneN == r) logits[eBase + mBase + r] = s + abv;
    }

    // ---- store messages (f16, coalesced via LDS transpose) ----
#pragma unroll
    for (int n = 0; n < 8; ++n)
#pragma unroll
      for (int r = 0; r < 8; ++r)
        hl[(r + mBase) * 128 + n * 16 + laneN] = (_Float16)acc2[n][r];
#pragma unroll
    for (int i = 0; i < 4; ++i) {
      int cid = i * 32 + lane;
      int m = cid >> 3, part = cid & 7;
      v16h v = *(const v16h*)(hl + m * 128 + part * 16);
      *(v16h*)(msg + (size_t)(eBase + m) * 128 + part * 16) = v;
    }
  }
}

// ---------------------------------------------------------------------------
// Kernels 2a/2b: global softmax over E logits (online max / sum-exp)
// ---------------------------------------------------------------------------
__global__ void __launch_bounds__(256)
softmax_part_kernel(const float* __restrict__ logits, int E,
                    float* __restrict__ partials) {
  float m = -3.4e38f, s = 0.f;
  for (int i = blockIdx.x * blockDim.x + threadIdx.x; i < E;
       i += gridDim.x * blockDim.x) {
    float x = logits[i];
    float nm = fmaxf(m, x);
    s = s * __expf(m - nm) + __expf(x - nm);
    m = nm;
  }
  __shared__ float sm[256], ss[256];
  sm[threadIdx.x] = m; ss[threadIdx.x] = s;
  __syncthreads();
  for (int off = 128; off > 0; off >>= 1) {
    if (threadIdx.x < off) {
      float m1 = sm[threadIdx.x], s1 = ss[threadIdx.x];
      float m2 = sm[threadIdx.x + off], s2 = ss[threadIdx.x + off];
      float nm = fmaxf(m1, m2);
      sm[threadIdx.x] = nm;
      ss[threadIdx.x] = s1 * __expf(m1 - nm) + s2 * __expf(m2 - nm);
    }
    __syncthreads();
  }
  if (threadIdx.x == 0) {
    partials[2 * blockIdx.x] = sm[0];
    partials[2 * blockIdx.x + 1] = ss[0];
  }
}

__global__ void __launch_bounds__(256)
softmax_final_kernel(const float* __restrict__ partials, int nParts,
                     float* __restrict__ MS) {
  float m = -3.4e38f, s = 0.f;
  for (int i = threadIdx.x; i < nParts; i += blockDim.x) {
    float m2 = partials[2 * i], s2 = partials[2 * i + 1];
    float nm = fmaxf(m, m2);
    s = s * __expf(m - nm) + s2 * __expf(m2 - nm);
    m = nm;
  }
  __shared__ float sm[256], ss[256];
  sm[threadIdx.x] = m; ss[threadIdx.x] = s;
  __syncthreads();
  for (int off = 128; off > 0; off >>= 1) {
    if (threadIdx.x < off) {
      float m1 = sm[threadIdx.x], s1 = ss[threadIdx.x];
      float m2 = sm[threadIdx.x + off], s2 = ss[threadIdx.x + off];
      float nm = fmaxf(m1, m2);
      sm[threadIdx.x] = nm;
      ss[threadIdx.x] = s1 * __expf(m1 - nm) + s2 * __expf(m2 - nm);
    }
    __syncthreads();
  }
  if (threadIdx.x == 0) { MS[0] = sm[0]; MS[1] = 1.0f / ss[0]; }
}

// ---------------------------------------------------------------------------
// Kernel 3: scale by softmax weight and scatter-add into agg (L2 atomics)
// ---------------------------------------------------------------------------
__global__ void __launch_bounds__(256)
scatter_kernel(const _Float16* __restrict__ msg, const float* __restrict__ logits,
               const int* __restrict__ ei, const float* __restrict__ MS,
               float* __restrict__ agg, int E, long long total) {
  const float M = MS[0], invS = MS[1];
  long long stride = (long long)gridDim.x * blockDim.x;
  for (long long i = (long long)blockIdx.x * blockDim.x + threadIdx.x; i < total;
       i += stride) {
    int e = (int)(i >> 7), d = (int)(i & 127);
    float w = __expf(logits[e] - M) * invS;
    float v = (float)msg[i] * w;
    atomicAdd(agg + (size_t)ei[E + e] * 128 + d, v);
  }
}

// ---------------------------------------------------------------------------
// Kernel 4: node update MLP ([nf | agg] -> 256x128 -> ReLU -> 128x128) + resid
// ---------------------------------------------------------------------------
__global__ void __launch_bounds__(256)
node_update_kernel(const float* __restrict__ nf, const float* __restrict__ agg,
                   const float* __restrict__ uW1, const float* __restrict__ ub1,
                   const float* __restrict__ uW2, const float* __restrict__ ub2,
                   float* __restrict__ out, int nTiles) {
  extern __shared__ char smem[];
  _Float16* w1f = (_Float16*)smem;          // 64 frags * 512
  _Float16* w2f = w1f + 64 * 512;           // 32 frags * 512
  _Float16* scratchAll = w2f + 32 * 512;    // 8 * 2048
  float* b1s = (float*)(scratchAll + 8 * 2048);
  float* b2s = b1s + 128;

  stage_wfrags(uW1, w1f, 64);
  stage_wfrags(uW2, w2f, 32);
  for (int i = threadIdx.x; i < 128; i += blockDim.x) {
    b1s[i] = ub1[i]; b2s[i] = ub2[i];
  }
  __syncthreads();

  const int lane  = threadIdx.x & 31;
  const int wave  = threadIdx.x >> 5;
  const int laneN = lane & 15;
  const int koff  = (lane < 16) ? 0 : 8;
  const int mBase = (lane < 16) ? 0 : 8;
  _Float16* hl = scratchAll + wave * 2048;

  int wgid = blockIdx.x * (blockDim.x >> 5) + wave;
  int wstride = gridDim.x * (blockDim.x >> 5);
  for (int t = wgid; t < nTiles; t += wstride) {
    const int vBase = t << 4;
    const int node = vBase + laneN;

    v8f acc[8];
#pragma unroll
    for (int n = 0; n < 8; ++n) {
      float bv = b1s[n * 16 + laneN];
#pragma unroll
      for (int r = 0; r < 8; ++r) acc[n][r] = bv;
    }
#pragma unroll
    for (int c = 0; c < 8; ++c) {
      const float* p = (c < 4)
          ? nf  + (size_t)node * 128 + c * 32 + koff
          : agg + (size_t)node * 128 + (c - 4) * 32 + koff;
      float4 x0 = *(const float4*)p;
      float4 x1 = *(const float4*)(p + 4);
      float4 x2 = *(const float4*)(p + 16);
      float4 x3 = *(const float4*)(p + 20);
      v16h a = pack16(x0, x1, x2, x3);
#pragma unroll
      for (int n = 0; n < 8; ++n) {
        v16h b = *(const v16h*)(w1f + ((c * 8 + n) << 9) + (lane << 4));
        acc[n] = __builtin_amdgcn_wmma_f32_16x16x32_f16(
            false, a, false, b, (short)0, acc[n], false, false);
      }
    }
#pragma unroll
    for (int n = 0; n < 8; ++n)
#pragma unroll
      for (int r = 0; r < 8; ++r) {
        float v = acc[n][r];
        hl[(r + mBase) * 128 + n * 16 + laneN] = (_Float16)(v > 0.f ? v : 0.f);
      }

    v8f acc2[8];
#pragma unroll
    for (int n = 0; n < 8; ++n) {
      float bv = b2s[n * 16 + laneN];
#pragma unroll
      for (int r = 0; r < 8; ++r) acc2[n][r] = bv;
    }
#pragma unroll
    for (int c = 0; c < 4; ++c) {
      v8h lo = *(const v8h*)(hl + laneN * 128 + c * 32 + koff);
      v8h hi = *(const v8h*)(hl + laneN * 128 + c * 32 + koff + 16);
      v16h a = __builtin_shufflevector(lo, hi, 0,1,2,3,4,5,6,7,8,9,10,11,12,13,14,15);
#pragma unroll
      for (int n = 0; n < 8; ++n) {
        v16h b = *(const v16h*)(w2f + ((c * 8 + n) << 9) + (lane << 4));
        acc2[n] = __builtin_amdgcn_wmma_f32_16x16x32_f16(
            false, a, false, b, (short)0, acc2[n], false, false);
      }
    }

    // residual add + f32 store
#pragma unroll
    for (int n = 0; n < 8; ++n)
#pragma unroll
      for (int r = 0; r < 8; ++r) {
        size_t idx = (size_t)(vBase + r + mBase) * 128 + n * 16 + laneN;
        out[idx] = acc2[n][r] + nf[idx];
      }
  }
}

// ---------------------------------------------------------------------------
// launch
// ---------------------------------------------------------------------------
extern "C" void kernel_launch(void* const* d_in, const int* in_sizes, int n_in,
                              void* d_out, int out_size, void* d_ws, size_t ws_size,
                              hipStream_t stream) {
  const float* nf  = (const float*)d_in[0];
  const float* ef  = (const float*)d_in[1];
  const float* mW1 = (const float*)d_in[2];
  const float* mb1 = (const float*)d_in[3];
  const float* mW2 = (const float*)d_in[4];
  const float* mb2 = (const float*)d_in[5];
  const float* aW  = (const float*)d_in[6];
  const float* ab  = (const float*)d_in[7];
  const float* uW1 = (const float*)d_in[8];
  const float* ub1 = (const float*)d_in[9];
  const float* uW2 = (const float*)d_in[10];
  const float* ub2 = (const float*)d_in[11];
  const int*   ei  = (const int*)d_in[12];

  const int E = in_sizes[12] / 2;
  const int N = in_sizes[0] / 128;
  float* out = (float*)d_out;

  // workspace layout
  char* ws = (char*)d_ws;
  _Float16* msg = (_Float16*)ws;                       // E*128 halves
  size_t off = (size_t)E * 128 * sizeof(_Float16);
  float* logits = (float*)(ws + off);  off += (size_t)E * sizeof(float);
  float* agg    = (float*)(ws + off);  off += (size_t)N * 128 * sizeof(float);
  float* partials = (float*)(ws + off); off += 2 * 512 * sizeof(float);
  float* MS     = (float*)(ws + off);

  // 0) zero aggregation buffer (harness does not re-zero ws between replays)
  zero_kernel<<<512, 256, 0, stream>>>((float4*)agg, (long long)N * 32);

  // 1) edge message MLP + logits
  const int eTiles = E / 16;
  size_t sm1 = (size_t)(72 * 512 + 32 * 512 + 8 * 2048) * sizeof(_Float16)
             + 3 * 128 * sizeof(float);
  edge_msg_kernel<<<512, 256, sm1, stream>>>(nf, ef, mW1, mb1, mW2, mb2,
                                             aW, ab, ei, msg, logits, E, eTiles);

  // 2) global softmax over edges
  softmax_part_kernel<<<512, 256, 0, stream>>>(logits, E, partials);
  softmax_final_kernel<<<1, 256, 0, stream>>>(partials, 512, MS);

  // 3) scale + scatter-add (agg is L2-resident: 51 MB << 192 MB)
  scatter_kernel<<<2048, 256, 0, stream>>>(msg, logits, ei, MS, agg, E,
                                           (long long)E * 128);

  // 4) node update MLP + residual
  const int nTiles = N / 16;
  size_t sm2 = (size_t)(64 * 512 + 32 * 512 + 8 * 2048) * sizeof(_Float16)
             + 2 * 128 * sizeof(float);
  node_update_kernel<<<512, 256, sm2, stream>>>(nf, agg, uW1, ub1, uW2, ub2,
                                                out, nTiles);
}
